// TemporalAttention_51926154609429
// MI455X (gfx1250) — matrix operand order repaired and checked by reading, not compiled, not run on hardware
//
#include <hip/hip_runtime.h>
#include <hip/hip_bf16.h>
#include <math.h>

typedef __bf16 v16bf __attribute__((ext_vector_type(16)));
typedef __bf16 v8bf  __attribute__((ext_vector_type(8)));
typedef float  v8f   __attribute__((ext_vector_type(8)));
typedef float  v2f   __attribute__((ext_vector_type(2)));

#define B_  16
#define T_  48
#define N_  256
#define D_  64
#define H_  8
#define HS_ 8

__device__ __forceinline__ float sigmoidf_(float x) {
    return 1.0f / (1.0f + __expf(-x));
}

// gfx1250 async global->LDS copy (per-lane gather), tracked by ASYNCcnt.
__device__ __forceinline__ void async_load_f32_to_lds(unsigned int lds_off,
                                                      const float* gptr) {
    asm volatile("global_load_async_to_lds_b32 %0, %1, off"
                 :: "v"(lds_off), "v"((unsigned long long)(uintptr_t)gptr)
                 : "memory");
}
__device__ __forceinline__ void wait_asynccnt0() {
    asm volatile("s_wait_asynccnt 0x0" ::: "memory");
}

// ---------------------------------------------------------------------------
// Kernel 1: per-(b,t) K/V projections.  k = key @ key_w + key_b  (256x64x64)
// Weights are async-gathered (transposed) into LDS once per block, converted
// to bf16 once; B fragments then come from single 32B LDS vector loads.
// ---------------------------------------------------------------------------
__global__ __launch_bounds__(256)
void kv_proj_kernel(const float* __restrict__ key, const float* __restrict__ value,
                    const float* __restrict__ key_w, const float* __restrict__ key_b,
                    const float* __restrict__ value_w, const float* __restrict__ value_b,
                    float* __restrict__ k_proj, float* __restrict__ v_proj)
{
    __shared__ float stage[2 * D_ * D_];                                   // 32 KB, w^T f32
    __shared__ __attribute__((aligned(32))) __bf16 wtb[2 * D_ * D_];       // 16 KB, w^T bf16

    const int tid  = threadIdx.x;
    const int wave = tid >> 5;
    const int lane = tid & 31;
    const int l = lane & 15, g = lane >> 4;
    const int bt = blockIdx.x;                    // 0 .. B*T-1

    const size_t srcBase = (size_t)bt * N_ * D_;
    const size_t wBase   = (size_t)bt * D_ * D_;
    const size_t bBase   = (size_t)bt * D_;

    // ---- async-stage key_w/value_w transposed into LDS: stage[m][e][k] = w[k][e]
    const unsigned int stageBase = (unsigned int)(uintptr_t)(void*)&stage[0];
    #pragma unroll
    for (int m = 0; m < 2; ++m) {
        const float* wg = (m ? value_w : key_w) + wBase;
        #pragma unroll
        for (int it = 0; it < 16; ++it) {
            const int i = tid + it * 256;         // 0..4095
            const int e = i & 63, k = i >> 6;     // lanes vary e -> coalesced reads
            async_load_f32_to_lds(stageBase + (unsigned int)(m * 4096 + e * 64 + k) * 4u,
                                  wg + (size_t)k * D_ + e);
        }
    }
    wait_asynccnt0();
    __syncthreads();
    #pragma unroll
    for (int it = 0; it < 32; ++it) {             // f32 -> bf16, once per block
        const int i = tid + it * 256;
        wtb[i] = (__bf16)stage[i];
    }
    __syncthreads();

    // ---- 128 output tiles (2 matrices x 16 Mtiles x 4 Ntiles) over 8 waves
    for (int tile = wave; tile < 128; tile += 8) {
        const int mat = tile >> 6;                // 0 = K, 1 = V
        const int mt  = (tile >> 2) & 15;
        const int nt  = tile & 3;
        const float* src = mat ? value   : key;
        const float* bia = mat ? value_b : key_b;
        float*       dst = mat ? v_proj  : k_proj;

        const int arow = mt * 16 + l;             // A row (token)
        const int ecol = nt * 16 + l;             // B/C column (channel)
        v8f c = {};
        #pragma unroll
        for (int kb = 0; kb < D_; kb += 32) {
            v16bf a;
            #pragma unroll
            for (int e = 0; e < 16; ++e) {        // A 16x32 bf16 layout
                const int ka = kb + g * 16 + e;
                a[e] = (__bf16)src[srcBase + (size_t)arow * D_ + ka];
            }
            // B 32x16 bf16: lane half g holds K = g*16 + e, contiguous in w^T
            const v16bf bm = *(const v16bf*)&wtb[mat * 4096 + ecol * 64 + kb + g * 16];
            c = __builtin_amdgcn_wmma_f32_16x16x32_bf16(false, a, false, bm,
                                                        (short)0, c, false, false);
        }
        const float bv = bia[bBase + ecol];
        #pragma unroll
        for (int r = 0; r < 8; ++r) {
            const int orow = mt * 16 + g * 8 + r;
            dst[srcBase + (size_t)orow * D_ + ecol] = c[r] + bv;
        }
    }
}

// ---------------------------------------------------------------------------
// Kernel 2: attention.  One wave per (b, n, h); 4 waves / block.
// S^T = K' x Q'^T via v_wmma_f32_16x16x4_f32 (softmax over keys becomes a
// column walk in LDS), elementwise decay/tanh + softmax, then out = att x V'.
// ---------------------------------------------------------------------------
__global__ __launch_bounds__(128)
void attn_kernel(const float* __restrict__ query,
                 const float* __restrict__ k_proj,
                 const float* __restrict__ v_proj,
                 const float* __restrict__ rate,
                 float* __restrict__ x_ws)
{
    __shared__ float sbuf[4 * T_ * T_];           // 36 KB: per-wave 48x48 S^T
    const int wave = threadIdx.x >> 5;
    const int lane = threadIdx.x & 31;
    const int l = lane & 15, g = lane >> 4;
    float* buf = &sbuf[wave * T_ * T_];

    const int idx = blockIdx.x * 4 + wave;        // 0 .. B*N*H-1
    const int h  = idx & 7;
    const int nn = (idx >> 3) & 255;
    const int b  = idx >> 11;

    const float sigrate   = sigmoidf_(rate[0]);
    const float inv_sq_hs = 0.35355339059327373f; // 1/sqrt(8)

    auto gaddr = [&](int p, int dk) -> size_t {   // [b, p, nn, h*8+dk]
        return ((size_t)(b * T_ + p) * N_ + nn) * (size_t)D_ + h * HS_ + dk;
    };

    // ----- S^T = K' x Q'^T : M = key pos, N = query pos, K = hs = 8 -----
    #pragma unroll
    for (int tm = 0; tm < 3; ++tm) {
        v2f a0, a1;
        #pragma unroll
        for (int j = 0; j < 2; ++j) {             // f32 A 16x4: K = 2g+j
            a0[j] = k_proj[gaddr(tm * 16 + l, 0 + 2 * g + j)];
            a1[j] = k_proj[gaddr(tm * 16 + l, 4 + 2 * g + j)];
        }
        #pragma unroll
        for (int tn = 0; tn < 3; ++tn) {
            v2f b0, b1;
            #pragma unroll
            for (int j = 0; j < 2; ++j) {         // B 4x16: col = lane, K = 2g+j
                b0[j] = query[gaddr(tn * 16 + l, 0 + 2 * g + j)];
                b1[j] = query[gaddr(tn * 16 + l, 4 + 2 * g + j)];
            }
            v8f c = {};
            c = __builtin_amdgcn_wmma_f32_16x16x4_f32(false, a0, false, b0,
                                                      (short)0, c, false, false);
            c = __builtin_amdgcn_wmma_f32_16x16x4_f32(false, a1, false, b1,
                                                      (short)0, c, false, false);
            #pragma unroll
            for (int r = 0; r < 8; ++r)
                buf[(tm * 16 + g * 8 + r) * T_ + tn * 16 + l] = c[r];
        }
    }
    __syncthreads();

    // ----- elementwise + softmax over keys (each lane owns column(s) q) -----
    #pragma unroll
    for (int pass = 0; pass < 2; ++pass) {
        const int q = lane + pass * 32;
        if (q < T_) {
            const float dec = (float)(T_ - q);
            float mx = -1e30f;
            for (int k = 0; k < T_; ++k) {
                const float s  = sigmoidf_(buf[k * T_ + q] * inv_sq_hs);
                const float dn = 1.0f + sigmoidf_(s * sigrate * dec);
                const float v  = tanhf(s / dn);
                buf[k * T_ + q] = v;
                mx = fmaxf(mx, v);
            }
            float sum = 0.0f;
            for (int k = 0; k < T_; ++k) {
                const float e = __expf(buf[k * T_ + q] - mx);
                buf[k * T_ + q] = e;
                sum += e;
            }
            const float inv = 1.0f / sum;
            for (int k = 0; k < T_; ++k)
                buf[k * T_ + q] *= inv;
        }
    }
    __syncthreads();

    // ----- out = att x V' : M = q (48), N = d (8 pad 16), K = key pos (48) -----
    v8f acc0 = {}, acc1 = {}, acc2 = {};
    #pragma unroll
    for (int kb = 0; kb < T_; kb += 4) {
        v2f bv;
        #pragma unroll
        for (int j = 0; j < 2; ++j)
            bv[j] = (l < HS_) ? v_proj[gaddr(kb + 2 * g + j, l)] : 0.0f;
        v2f a0, a1, a2;
        #pragma unroll
        for (int j = 0; j < 2; ++j) {             // att[q,k] = buf[k*48 + q]
            a0[j] = buf[(kb + 2 * g + j) * T_ +  0 + l];
            a1[j] = buf[(kb + 2 * g + j) * T_ + 16 + l];
            a2[j] = buf[(kb + 2 * g + j) * T_ + 32 + l];
        }
        acc0 = __builtin_amdgcn_wmma_f32_16x16x4_f32(false, a0, false, bv,
                                                     (short)0, acc0, false, false);
        acc1 = __builtin_amdgcn_wmma_f32_16x16x4_f32(false, a1, false, bv,
                                                     (short)0, acc1, false, false);
        acc2 = __builtin_amdgcn_wmma_f32_16x16x4_f32(false, a2, false, bv,
                                                     (short)0, acc2, false, false);
    }
    if (l < HS_) {
        #pragma unroll
        for (int r = 0; r < 8; ++r) {
            x_ws[gaddr( 0 + g * 8 + r, l)] = acc0[r];
            x_ws[gaddr(16 + g * 8 + r, l)] = acc1[r];
            x_ws[gaddr(32 + g * 8 + r, l)] = acc2[r];
        }
    }
}

// ---------------------------------------------------------------------------
// Kernel 3: out = tanh(x @ P1^T + b1) @ P2^T + b2.  128 rows / block.
// P1/P2 staged once as bf16 in LDS (already contiguous in the K direction);
// intermediate h stored bf16 so GEMM2 fragments are pure vector ds loads.
// ---------------------------------------------------------------------------
__global__ __launch_bounds__(256)
void proj_kernel(const float* __restrict__ x,
                 const float* __restrict__ p1w, const float* __restrict__ p1b,
                 const float* __restrict__ p2w, const float* __restrict__ p2b,
                 float* __restrict__ out)
{
    __shared__ __attribute__((aligned(32))) __bf16 w1b[D_ * D_];   //  8 KB
    __shared__ __attribute__((aligned(32))) __bf16 w2b[D_ * D_];   //  8 KB
    __shared__ __attribute__((aligned(32))) __bf16 hbuf[128 * D_]; // 16 KB
    const int tid  = threadIdx.x;
    const int wave = tid >> 5;                    // = local M tile (8 tiles)
    const int lane = tid & 31;
    const int l = lane & 15, g = lane >> 4;
    const size_t rowBase = (size_t)blockIdx.x * 128;
    const int arow = wave * 16 + l;

    // stage both weight matrices as bf16 (B[dk,e] = P[e*64+dk]: rows already
    // contiguous over dk, no transpose needed)
    #pragma unroll
    for (int it = 0; it < 16; ++it) {
        const int i = tid + it * 256;             // 0..4095
        w1b[i] = (__bf16)p1w[i];
        w2b[i] = (__bf16)p2w[i];
    }
    __syncthreads();

    // GEMM1: h = tanh(x @ P1^T + b1)
    #pragma unroll
    for (int nt = 0; nt < 4; ++nt) {
        const int ecol = nt * 16 + l;
        v8f c = {};
        #pragma unroll
        for (int kb = 0; kb < D_; kb += 32) {
            v16bf a;
            #pragma unroll
            for (int e = 0; e < 16; ++e) {
                const int ka = kb + g * 16 + e;
                a[e] = (__bf16)x[(rowBase + arow) * D_ + ka];
            }
            const v16bf bm = *(const v16bf*)&w1b[ecol * D_ + kb + g * 16];
            c = __builtin_amdgcn_wmma_f32_16x16x32_bf16(false, a, false, bm,
                                                        (short)0, c, false, false);
        }
        const float bv = p1b[ecol];
        #pragma unroll
        for (int r = 0; r < 8; ++r)
            hbuf[(wave * 16 + g * 8 + r) * D_ + ecol] = (__bf16)tanhf(c[r] + bv);
    }
    __syncthreads();

    // GEMM2: out = h @ P2^T + b2  (A fragments: two 16B LDS vector loads)
    #pragma unroll
    for (int nt = 0; nt < 4; ++nt) {
        const int ecol = nt * 16 + l;
        v8f c = {};
        #pragma unroll
        for (int kb = 0; kb < D_; kb += 32) {
            const v8bf lo = *(const v8bf*)&hbuf[arow * D_ + kb + g * 16];
            const v8bf hi = *(const v8bf*)&hbuf[arow * D_ + kb + g * 16 + 8];
            v16bf a;
            #pragma unroll
            for (int e = 0; e < 8; ++e) { a[e] = lo[e]; a[8 + e] = hi[e]; }
            const v16bf bm = *(const v16bf*)&w2b[ecol * D_ + kb + g * 16];
            c = __builtin_amdgcn_wmma_f32_16x16x32_bf16(false, a, false, bm,
                                                        (short)0, c, false, false);
        }
        const float bv = p2b[ecol];
        #pragma unroll
        for (int r = 0; r < 8; ++r)
            out[(rowBase + wave * 16 + g * 8 + r) * D_ + ecol] = c[r] + bv;
    }
}

// ---------------------------------------------------------------------------
extern "C" void kernel_launch(void* const* d_in, const int* in_sizes, int n_in,
                              void* d_out, int out_size, void* d_ws, size_t ws_size,
                              hipStream_t stream) {
    const float* query   = (const float*)d_in[0];
    const float* key     = (const float*)d_in[1];
    const float* value   = (const float*)d_in[2];
    const float* key_w   = (const float*)d_in[3];
    const float* key_b   = (const float*)d_in[4];
    const float* value_w = (const float*)d_in[5];
    const float* value_b = (const float*)d_in[6];
    const float* p1w     = (const float*)d_in[7];
    const float* p1b     = (const float*)d_in[8];
    const float* p2w     = (const float*)d_in[9];
    const float* p2b     = (const float*)d_in[10];
    const float* rate    = (const float*)d_in[11];
    float* out = (float*)d_out;

    const size_t elems = (size_t)B_ * T_ * N_ * D_;   // 12,582,912
    float* ws     = (float*)d_ws;
    float* k_proj = ws;
    float* v_proj = ws + elems;
    float* x_ws   = ws + 2 * elems;

    kv_proj_kernel<<<B_ * T_, 256, 0, stream>>>(key, value, key_w, key_b,
                                                value_w, value_b, k_proj, v_proj);
    attn_kernel<<<(B_ * N_ * H_) / 4, 128, 0, stream>>>(query, k_proj, v_proj,
                                                        rate, x_ws);
    proj_kernel<<<(int)((size_t)B_ * T_ * N_ / 128), 256, 0, stream>>>(
        x_ws, p1w, p1b, p2w, p2b, out);
}
